// ProofGenerator_37864431681873
// MI455X (gfx1250) — compile-verified, matrix operand order checked by reading, not run on hardware
//
#include <hip/hip_runtime.h>
#include <hip/hip_bf16.h>
#include <stdint.h>

typedef unsigned short u16;
typedef unsigned int   u32;
typedef __attribute__((ext_vector_type(16))) __bf16 v16bf;
typedef __attribute__((ext_vector_type(8)))  float  v8f;

// ---------- small device helpers ----------
__device__ inline u16 f2bf(float f) {
    u32 x = __float_as_uint(f);
    u32 r = x + 0x7FFFu + ((x >> 16) & 1u);   // round-to-nearest-even
    return (u16)(r >> 16);
}
__device__ inline float bf2f(u16 h) { return __uint_as_float(((u32)h) << 16); }
__device__ inline float geluf(float x) { return 0.5f * x * (1.0f + erff(x * 0.70710678118654752440f)); }
__device__ inline float sigmf(float x) { return 1.0f / (1.0f + expf(-x)); }

// ---------- elementwise / utility kernels ----------
__global__ void k_f32_to_bf16(const float* __restrict__ s, u16* __restrict__ d, int n) {
    int i = blockIdx.x * 256 + threadIdx.x;
    if (i < n) d[i] = f2bf(s[i]);
}

// rule_emb (32x1024) -> transposed bf16 (1024x32) so the standard A@W^T GEMM applies
__global__ void k_embT(const float* __restrict__ s, u16* __restrict__ d) {
    int i = blockIdx.x * 256 + threadIdx.x;
    if (i < 32 * 1024) { int r = i >> 10, h = i & 1023; d[h * 32 + r] = f2bf(s[i]); }
}

__global__ void k_zero_f32(float* p, int n) { int i = blockIdx.x * 256 + threadIdx.x; if (i < n) p[i] = 0.0f; }
__global__ void k_zero_u16(u16* p, int n)   { int i = blockIdx.x * 256 + threadIdx.x; if (i < n) p[i] = 0; }

// proof_steps[0] = comp[:, 0, :]
__global__ void k_copy_ps0(const float* __restrict__ comp, float* __restrict__ dst) {
    int i = blockIdx.x * 256 + threadIdx.x;
    if (i < 32 * 1024) { int b = i >> 10, h = i & 1023; dst[i] = comp[((size_t)b << 20) + h]; }
}

// ---------- WMMA bf16 GEMM:  out = act(A @ W^T + bias + add) * outscale ----------
// A: M x K bf16 row-major (activations), W: N x K bf16 row-major (torch Linear weight)
// 2 waves along M (tile 32 rows); WN waves along N; each wave owns SUBS 16-wide
// subtiles. Tile geometry is compile-time => NO runtime guards => no EXEC
// divergence around WMMA (ISA requires EXEC all-ones) and accumulators stay
// pinned in fixed VGPRs (no phi-node mov storms).
template <int ACT, int WN, int SUBS>
__global__ __launch_bounds__(64 * WN)
void k_gemm(const u16* __restrict__ A, const u16* __restrict__ W,
            const float* __restrict__ bias, const float* __restrict__ add,
            float* __restrict__ outF, u16* __restrict__ outB,
            int M, int N, int K, float outscale) {
    constexpr int BT = 64 * WN;                 // block threads (2*WN waves)
    constexpr int NT = WN * SUBS * 16;          // block tile width in N
    __shared__ u16 As[32 * 32];                 // 2 KB A slab, [m][k]
    const int tid   = threadIdx.x;
    const int lane  = tid & 31;
    const int wave  = tid >> 5;
    const int waveM = wave / WN;                // 0..1
    const int waveN = wave % WN;                // 0..WN-1
    const int m0    = blockIdx.y * 32;
    const int n0    = blockIdx.x * NT + waveN * (SUBS * 16);
    const int half  = lane >> 4;                // 0..1
    const int lmod  = lane & 15;
    const int kbase = half * 8;

    v8f acc[SUBS];
#pragma unroll
    for (int i = 0; i < SUBS; ++i)
#pragma unroll
        for (int j = 0; j < 8; ++j) acc[i][j] = 0.0f;

    for (int k0 = 0; k0 < K; k0 += 32) {
        // cooperative A slab load: 32x32 bf16 = 128 uint4
        for (int idx = tid; idx < 128; idx += BT) {
            int r = idx >> 2, c = (idx & 3) * 8;
            *(uint4*)(&As[r * 32 + c]) = *(const uint4*)(&A[(size_t)(m0 + r) * K + k0 + c]);
        }
        __syncthreads();

        union { v16bf v; uint4 q[2]; } afrag;   // 16-bit A-frag VGPR layout (ISA 7.12.2)
        {
            int r = waveM * 16 + lmod;
            afrag.q[0] = *(const uint4*)(&As[r * 32 + kbase]);
            afrag.q[1] = *(const uint4*)(&As[r * 32 + 16 + kbase]);
        }

        if (k0 + 32 < K)                        // global_prefetch_b8 of next K-slab of W
            __builtin_prefetch(&W[(size_t)(n0 + lmod) * K + k0 + 32], 0, 1);

#pragma unroll
        for (int sub = 0; sub < SUBS; ++sub) {
            union { v16bf v; uint4 q[2]; } bfrag;
            const u16* wr = &W[(size_t)(n0 + sub * 16 + lmod) * K + k0];
            bfrag.q[0] = *(const uint4*)(wr + kbase);
            bfrag.q[1] = *(const uint4*)(wr + 16 + kbase);
            acc[sub] = __builtin_amdgcn_wmma_f32_16x16x32_bf16(
                false, afrag.v, false, bfrag.v, (short)0, acc[sub], false, false);
        }
        __syncthreads();
    }

#pragma unroll
    for (int sub = 0; sub < SUBS; ++sub) {
        int col = n0 + sub * 16 + lmod;
        float bv = bias ? bias[col] : 0.0f;
#pragma unroll
        for (int j = 0; j < 8; ++j) {           // C-frag: VGPR j -> rows j / j+8 (ISA 7.12.2)
            int row   = m0 + waveM * 16 + half * 8 + j;
            size_t ix = (size_t)row * N + col;
            float f = acc[sub][j] + bv;
            if (add) f += add[ix];
            if (ACT == 1) f = geluf(f);
            f *= outscale;
            if (outF) outF[ix] = f;
            if (outB) outB[ix] = f2bf(f);
        }
    }
}

// ---------- rule softmax over R=32 (one wave per row) ----------
__global__ void k_softmax32(const float* __restrict__ logits,
                            float* __restrict__ pout, u16* __restrict__ pbf) {
    int row = blockIdx.x, lane = threadIdx.x;
    float v = logits[row * 32 + lane];
    float m = v;
    for (int off = 16; off > 0; off >>= 1) m = fmaxf(m, __shfl_xor(m, off, 32));
    float e = expf(v - m), s = e;
    for (int off = 16; off > 0; off >>= 1) s += __shfl_xor(s, off, 32);
    float p = e / s;
    pout[row * 32 + lane] = p;
    pbf[row * 32 + lane]  = f2bf(p);
}

// ---------- attention: one block per (head, batch) ----------
__global__ __launch_bounds__(256)
void k_attn(const float* __restrict__ q, const u16* __restrict__ Kb, const u16* __restrict__ Vb,
            float* __restrict__ ctxF, u16* __restrict__ ctxB, float* __restrict__ attn_acc) {
    const int h = blockIdx.x, b = blockIdx.y, tid = threadIdx.x;
    __shared__ float qs[64];
    __shared__ float sc[1024];
    __shared__ float red[256];
    if (tid < 64) qs[tid] = q[b * 1024 + h * 64 + tid];
    __syncthreads();

    float lmax = -1e30f;
    for (int s = tid; s < 1024; s += 256) {
        const u16* kr = Kb + ((((size_t)b * 1024 + s) * 16 + h) << 6);
        float a = 0.0f;
        for (int d0 = 0; d0 < 64; ++d0) a += qs[d0] * bf2f(kr[d0]);
        sc[s] = a;
        lmax = fmaxf(lmax, a);
    }
    red[tid] = lmax; __syncthreads();
    for (int st = 128; st > 0; st >>= 1) { if (tid < st) red[tid] = fmaxf(red[tid], red[tid + st]); __syncthreads(); }
    float m = red[0]; __syncthreads();

    float lsum = 0.0f;
    for (int s = tid; s < 1024; s += 256) { float e = expf(sc[s] - m); sc[s] = e; lsum += e; }
    red[tid] = lsum; __syncthreads();
    for (int st = 128; st > 0; st >>= 1) { if (tid < st) red[tid] += red[tid + st]; __syncthreads(); }
    float inv = 1.0f / red[0]; __syncthreads();

    for (int s = tid; s < 1024; s += 256)       // attn_avg = mean over 16 heads
        atomicAdd(&attn_acc[b * 1024 + s], sc[s] * inv * (1.0f / 16.0f));

    int d = tid & 63, part = tid >> 6;
    float a = 0.0f;
    for (int s = part * 256; s < part * 256 + 256; ++s)
        a += sc[s] * bf2f(Vb[((((size_t)b * 1024 + s) * 16 + h) << 6) + d]);
    red[tid] = a * inv; __syncthreads();
    if (tid < 64) {
        float r = red[tid] + red[tid + 64] + red[tid + 128] + red[tid + 192];
        int ix = b * 1024 + h * 64 + tid;
        ctxF[ix] = r; ctxB[ix] = f2bf(r);
    }
}

// ---------- LSTM cell (gates pre-summed), in-place c update ----------
__global__ void k_lstm_cell(const float* __restrict__ g, float* __restrict__ c,
                            float* __restrict__ h, u16* __restrict__ hbf) {
    int i = blockIdx.x * 256 + threadIdx.x;
    if (i >= 32 * 1024) return;
    int b = i >> 10, j = i & 1023;
    size_t base = (size_t)b * 4096 + j;
    float gi = g[base], gf = g[base + 1024], gg = g[base + 2048], go = g[base + 3072];
    float cn = sigmf(gf) * c[i] + sigmf(gi) * tanhf(gg);
    float hn = sigmf(go) * tanhf(cn);
    c[i] = cn; h[i] = hn; hbf[i] = f2bf(hn);
}

__global__ void k_add4_bf(const float* __restrict__ a, const float* __restrict__ b,
                          const float* __restrict__ c, const float* __restrict__ d,
                          u16* __restrict__ o, int n) {
    int i = blockIdx.x * 256 + threadIdx.x;
    if (i < n) o[i] = f2bf(a[i] + b[i] + c[i] + d[i]);
}

// ---------- layernorm(2048) + GELU -> bf16 ----------
__global__ __launch_bounds__(256)
void k_ln_gelu(const float* __restrict__ z, const float* __restrict__ gm,
               const float* __restrict__ bt, u16* __restrict__ o) {
    int row = blockIdx.x, tid = threadIdx.x;
    const float* zr = z + (size_t)row * 2048;
    __shared__ float s1[256], s2[256];
    float a = 0.0f, a2 = 0.0f;
    for (int i = tid; i < 2048; i += 256) { float v = zr[i]; a += v; a2 += v * v; }
    s1[tid] = a; s2[tid] = a2; __syncthreads();
    for (int st = 128; st > 0; st >>= 1) {
        if (tid < st) { s1[tid] += s1[tid + st]; s2[tid] += s2[tid + st]; }
        __syncthreads();
    }
    float mu  = s1[0] * (1.0f / 2048.0f);
    float var = s2[0] * (1.0f / 2048.0f) - mu * mu;
    float rs  = rsqrtf(var + 1e-5f);
    for (int i = tid; i < 2048; i += 256) {
        float v = (zr[i] - mu) * rs * gm[i] + bt[i];
        o[(size_t)row * 2048 + i] = f2bf(geluf(v));
    }
}

__global__ void k_concat(const u16* __restrict__ a, const u16* __restrict__ b, u16* __restrict__ o) {
    int i = blockIdx.x * 256 + threadIdx.x;
    if (i < 32 * 2048) {
        int r = i >> 11, j = i & 2047;
        o[i] = (j < 1024) ? a[r * 1024 + j] : b[r * 1024 + (j - 1024)];
    }
}

// sigmoid(x @ w(1xH).T + b): one block per batch row
__global__ __launch_bounds__(256)
void k_head(const u16* __restrict__ x, const float* __restrict__ w,
            const float* __restrict__ bs, float* __restrict__ out) {
    int b = blockIdx.x, tid = threadIdx.x;
    __shared__ float r[256];
    float a = 0.0f;
    for (int i = tid; i < 1024; i += 256) a += bf2f(x[b * 1024 + i]) * w[i];
    r[tid] = a; __syncthreads();
    for (int st = 128; st > 0; st >>= 1) { if (tid < st) r[tid] += r[tid + st]; __syncthreads(); }
    if (tid == 0) out[b] = sigmf(r[0] + bs[0]);
}

// ---------------------------------------------------------------------------
extern "C" void kernel_launch(void* const* d_in, const int* in_sizes, int n_in,
                              void* d_out, int out_size, void* d_ws, size_t ws_size,
                              hipStream_t stream) {
    (void)in_sizes; (void)n_in; (void)out_size; (void)ws_size;
    const int B = 32, H = 1024, STEPS = 16;

    const float* comp       = (const float*)d_in[0];
    // d_in[1] = max_steps (16; must be graph-static)
    const float* rule_w1    = (const float*)d_in[2];
    const float* rule_b1    = (const float*)d_in[3];
    const float* rule_w2    = (const float*)d_in[4];
    const float* rule_b2    = (const float*)d_in[5];
    const float* rule_emb   = (const float*)d_in[6];
    const float* attn_in_w  = (const float*)d_in[7];
    const float* attn_in_b  = (const float*)d_in[8];
    const float* attn_out_w = (const float*)d_in[9];
    const float* attn_out_b = (const float*)d_in[10];
    const float* lstm_wih0  = (const float*)d_in[11];
    const float* lstm_whh0  = (const float*)d_in[12];
    const float* lstm_bih0  = (const float*)d_in[13];
    const float* lstm_bhh0  = (const float*)d_in[14];
    const float* lstm_wih1  = (const float*)d_in[15];
    const float* lstm_whh1  = (const float*)d_in[16];
    const float* lstm_bih1  = (const float*)d_in[17];
    const float* lstm_bhh1  = (const float*)d_in[18];
    const float* gen_w1     = (const float*)d_in[19];
    const float* gen_b1     = (const float*)d_in[20];
    const float* gen_ln_g   = (const float*)d_in[21];
    const float* gen_ln_b   = (const float*)d_in[22];
    const float* gen_w2     = (const float*)d_in[23];
    const float* gen_b2     = (const float*)d_in[24];
    const float* val_w1     = (const float*)d_in[25];
    const float* val_b1     = (const float*)d_in[26];
    const float* val_w2     = (const float*)d_in[27];
    const float* val_b2     = (const float*)d_in[28];
    const float* ver_w1     = (const float*)d_in[29];
    const float* ver_b1     = (const float*)d_in[30];
    const float* ver_w2     = (const float*)d_in[31];
    const float* ver_b2     = (const float*)d_in[32];

    // output slices (return order: steps, validity, probs, vals, attn)
    float* out     = (float*)d_out;
    float* o_steps = out;                 // 17*32*1024
    float* o_valid = out + 557056;        // 32
    float* o_probs = out + 557088;        // 16*32*32
    float* o_vals  = out + 573472;        // 16*32
    float* o_attn  = out + 573984;        // 16*32*1024

    // workspace bump allocator
    char* ws = (char*)d_ws;
    size_t off = 0;
    auto ALLOC = [&](size_t bytes) -> void* {
        void* p = ws + off;
        off = (off + bytes + 255) & ~(size_t)255;
        return p;
    };

    // bf16 weights (~56 MB, L2-resident across all 16 iterations)
    u16* rw1b   = (u16*)ALLOC((size_t)1048576 * 2);
    u16* rw2b   = (u16*)ALLOC((size_t)32768 * 2);
    u16* embTb  = (u16*)ALLOC((size_t)32768 * 2);
    u16* aiwb   = (u16*)ALLOC((size_t)3145728 * 2);
    u16* aowb   = (u16*)ALLOC((size_t)1048576 * 2);
    u16* wih0b  = (u16*)ALLOC((size_t)4194304 * 2);
    u16* whh0b  = (u16*)ALLOC((size_t)4194304 * 2);
    u16* wih1b  = (u16*)ALLOC((size_t)4194304 * 2);
    u16* whh1b  = (u16*)ALLOC((size_t)4194304 * 2);
    u16* gw1b   = (u16*)ALLOC((size_t)2097152 * 2);
    u16* gw2b   = (u16*)ALLOC((size_t)2097152 * 2);
    u16* vw1b   = (u16*)ALLOC((size_t)2097152 * 2);
    u16* verw1b = (u16*)ALLOC((size_t)1048576 * 2);
    // large tensors
    u16* compb  = (u16*)ALLOC((size_t)33554432 * 2);
    u16* Kbp    = (u16*)ALLOC((size_t)33554432 * 2);
    u16* Vbp    = (u16*)ALLOC((size_t)33554432 * 2);
    // per-step activations
    u16*   ps_bf    = (u16*)ALLOC(32768 * 2);
    u16*   t1_bf    = (u16*)ALLOC(32768 * 2);
    u16*   probs_bf = (u16*)ALLOC(1024 * 2);
    float* logits_f = (float*)ALLOC(1024 * 4);
    float* sel_f    = (float*)ALLOC(32768 * 4);
    float* q_f      = (float*)ALLOC(32768 * 4);
    float* ctx_f    = (float*)ALLOC(32768 * 4);
    u16*   ctx_bf   = (u16*)ALLOC(32768 * 2);
    float* ctxo_f   = (float*)ALLOC(32768 * 4);
    float* gA       = (float*)ALLOC(131072 * 4);
    float* gB       = (float*)ALLOC(131072 * 4);
    float* h0 = (float*)ALLOC(32768 * 4); float* c0 = (float*)ALLOC(32768 * 4);
    float* h1 = (float*)ALLOC(32768 * 4); float* c1 = (float*)ALLOC(32768 * 4);
    u16* h0b = (u16*)ALLOC(32768 * 2);   u16* h1b = (u16*)ALLOC(32768 * 2);
    u16* si_bf  = (u16*)ALLOC(32768 * 2);
    float* z_f  = (float*)ALLOC(65536 * 4);
    u16* zg_bf  = (u16*)ALLOC(65536 * 2);
    u16* nxt_bf = (u16*)ALLOC(32768 * 2);
    u16* cat_bf = (u16*)ALLOC(65536 * 2);
    u16* v1_bf  = (u16*)ALLOC(32768 * 2);
    u16* ver_bf = (u16*)ALLOC(32768 * 2);

    auto CVT = [&](const float* s, u16* d, int n) {
        k_f32_to_bf16<<<(n + 255) / 256, 256, 0, stream>>>(s, d, n);
    };
    // dispatch: N % 256 == 0 -> 32x256 tile (8 waves); N == 32 -> 32x32 tile (2 waves)
    auto GEMM = [&](int act, const u16* A, const u16* Wm, const float* bias, const float* add,
                    float* oF, u16* oB, int M, int N, int K, float sc) {
        if ((N & 255) == 0) {
            dim3 g(N / 256, M / 32);
            if (act) k_gemm<1, 4, 4><<<g, 256, 0, stream>>>(A, Wm, bias, add, oF, oB, M, N, K, sc);
            else     k_gemm<0, 4, 4><<<g, 256, 0, stream>>>(A, Wm, bias, add, oF, oB, M, N, K, sc);
        } else {                                 // N == 32
            dim3 g(N / 32, M / 32);
            if (act) k_gemm<1, 1, 2><<<g, 64, 0, stream>>>(A, Wm, bias, add, oF, oB, M, N, K, sc);
            else     k_gemm<0, 1, 2><<<g, 64, 0, stream>>>(A, Wm, bias, add, oF, oB, M, N, K, sc);
        }
    };

    // ---- one-time conversions ----
    CVT(rule_w1, rw1b, 1048576);
    CVT(rule_w2, rw2b, 32768);
    k_embT<<<128, 256, 0, stream>>>(rule_emb, embTb);
    CVT(attn_in_w, aiwb, 3145728);
    CVT(attn_out_w, aowb, 1048576);
    CVT(lstm_wih0, wih0b, 4194304); CVT(lstm_whh0, whh0b, 4194304);
    CVT(lstm_wih1, wih1b, 4194304); CVT(lstm_whh1, whh1b, 4194304);
    CVT(gen_w1, gw1b, 2097152);
    CVT(gen_w2, gw2b, 2097152);
    CVT(val_w1, vw1b, 2097152);
    CVT(ver_w1, verw1b, 1048576);
    CVT(comp, compb, 33554432);

    k_copy_ps0<<<128, 256, 0, stream>>>(comp, o_steps);
    k_zero_f32<<<128, 256, 0, stream>>>(c0, 32768);
    k_zero_f32<<<128, 256, 0, stream>>>(c1, 32768);
    k_zero_f32<<<128, 256, 0, stream>>>(h0, 32768);
    k_zero_f32<<<128, 256, 0, stream>>>(h1, 32768);
    k_zero_u16<<<128, 256, 0, stream>>>(h0b, 32768);
    k_zero_u16<<<128, 256, 0, stream>>>(h1b, 32768);

    // ---- K/V precompute (the FLOP-dominant WMMA GEMMs) ----
    GEMM(0, compb, aiwb + (size_t)1048576, attn_in_b + 1024, nullptr, nullptr, Kbp, 32768, 1024, 1024, 1.0f);
    GEMM(0, compb, aiwb + (size_t)2097152, attn_in_b + 2048, nullptr, nullptr, Vbp, 32768, 1024, 1024, 1.0f);

    // ---- recurrent loop ----
    for (int step = 0; step < STEPS; ++step) {
        float* ps   = o_steps + (size_t)step * 32768;
        float* nxtF = o_steps + (size_t)(step + 1) * 32768;
        float* prob = o_probs + (size_t)step * 1024;
        float* valp = o_vals + (size_t)step * 32;
        float* attn = o_attn + (size_t)step * 32768;

        CVT(ps, ps_bf, 32768);

        // rule selector
        GEMM(1, ps_bf, rw1b, rule_b1, nullptr, nullptr, t1_bf, B, H, H, 1.0f);
        GEMM(0, t1_bf, rw2b, rule_b2, nullptr, logits_f, nullptr, B, 32, H, 1.0f);
        k_softmax32<<<32, 32, 0, stream>>>(logits_f, prob, probs_bf);
        GEMM(0, probs_bf, embTb, nullptr, nullptr, sel_f, nullptr, B, H, 32, 1.0f);

        // attention
        GEMM(0, ps_bf, aiwb, attn_in_b, nullptr, q_f, nullptr, B, H, H, 0.125f); // q * dh^-0.5
        k_zero_f32<<<128, 256, 0, stream>>>(attn, 32768);
        k_attn<<<dim3(16, 32), 256, 0, stream>>>(q_f, Kbp, Vbp, ctx_f, ctx_bf, attn);
        GEMM(0, ctx_bf, aowb, attn_out_b, nullptr, ctxo_f, nullptr, B, H, H, 1.0f);

        // 2-layer LSTM
        GEMM(0, ps_bf, wih0b, lstm_bih0, nullptr, gA, nullptr, B, 4096, H, 1.0f);
        GEMM(0, h0b,   whh0b, lstm_bhh0, gA,      gB, nullptr, B, 4096, H, 1.0f);
        k_lstm_cell<<<128, 256, 0, stream>>>(gB, c0, h0, h0b);
        GEMM(0, h0b, wih1b, lstm_bih1, nullptr, gA, nullptr, B, 4096, H, 1.0f);
        GEMM(0, h1b, whh1b, lstm_bhh1, gA,      gB, nullptr, B, 4096, H, 1.0f);
        k_lstm_cell<<<128, 256, 0, stream>>>(gB, c1, h1, h1b);

        // generator
        k_add4_bf<<<128, 256, 0, stream>>>(ps, sel_f, ctxo_f, h1, si_bf, 32768);
        GEMM(0, si_bf, gw1b, gen_b1, nullptr, z_f, nullptr, B, 2048, H, 1.0f);
        k_ln_gelu<<<32, 256, 0, stream>>>(z_f, gen_ln_g, gen_ln_b, zg_bf);
        GEMM(0, zg_bf, gw2b, gen_b2, nullptr, nxtF, nxt_bf, B, H, 2048, 1.0f);

        // value head
        k_concat<<<256, 256, 0, stream>>>(ps_bf, nxt_bf, cat_bf);
        GEMM(1, cat_bf, vw1b, val_b1, nullptr, nullptr, v1_bf, B, H, 2048, 1.0f);
        k_head<<<32, 256, 0, stream>>>(v1_bf, val_w2, val_b2, valp);
    }

    // ---- verifier on final state ----
    GEMM(1, nxt_bf, verw1b, ver_b1, nullptr, nullptr, ver_bf, B, H, H, 1.0f);
    k_head<<<32, 256, 0, stream>>>(ver_bf, ver_w2, ver_b2, o_valid);
}
